// RQ_VAE_Quantizer_49005576847517
// MI455X (gfx1250) — matrix operand order repaired and checked by reading, not compiled, not run on hardware
//
#include <hip/hip_runtime.h>

// ---------------------------------------------------------------------------
// RQ-VAE residual quantizer for MI455X (gfx1250, wave32, WMMA).
//
// Single-pass: z read once, residual carried in registers across all 3
// layers. Distance GEMM r.W^T on the matrix units (v_wmma_f32_16x16x32_f16,
// 16 KFLOP/instr, 48 per wave). Argmin uses biased-positive distances packed
// with the codeword index into a uint32 key (monotone order) -> the compiler
// emits v_pk_fma_f32 + v_and_or_b32 + v_min_u32 per candidate group, then a
// per-row LDS ds_min_u32 atomic tournament (DS pipe, runs alongside VALU)
// replaces shuffle reductions and the index-ownership exchange. Residual
// update/accumulation stays in f32 (the numerically sensitive chain), with
// codewords fetched from the f32 LDS copy.
//
// Roofline: one pass moves ~134 MB (z in, quantized + indices out) ->
// ~5.8 us at 23.3 TB/s; the f16 WMMA covers the 25.8 GFLOP distance GEMM in
// ~1.57M matrix instructions, and the packed-key argmin keeps the remaining
// VALU work within ~2x of the memory time.
// ---------------------------------------------------------------------------

typedef __attribute__((ext_vector_type(16))) _Float16 v16h;
typedef __attribute__((ext_vector_type(8)))  _Float16 v8h;
typedef __attribute__((ext_vector_type(8)))  float    v8f;

#define LAYERS 3
#define KCW 256
#define DIM 32
#define ROWS_PER_WAVE 16
#define WAVES_PER_WG 8
#define ROWS_PER_WG (ROWS_PER_WAVE * WAVES_PER_WG)
#define DIST_BIAS 4.0f   // makes dist = bias + ||W||^2 - 2 r.W provably > 0

// LDS carve-up (bytes)
#define OFF_CWF 0
#define SZ_CWF  (LAYERS * KCW * DIM * 4)              // 98304 : f32 codebooks
#define OFF_CWH (OFF_CWF + SZ_CWF)
#define SZ_CWH  (LAYERS * KCW * DIM * 2)              // 49152 : f16 codebooks
#define OFF_WSQ (OFF_CWH + SZ_CWH)
#define SZ_WSQ  (LAYERS * KCW * 4)                    //  3072 : ||W||^2 + bias
#define OFF_IDX (OFF_WSQ + SZ_WSQ)
#define SZ_IDX  (WAVES_PER_WG * ROWS_PER_WAVE * 4)    //   512 : per-wave key mins
#define SMEM_BYTES (OFF_IDX + SZ_IDX)

__global__ __launch_bounds__(256) void rq_vae_wmma_kernel(
    const float* __restrict__ z,        // [N, 32]
    const float* __restrict__ cb,       // [3, 256, 32]
    float* __restrict__ outq,           // [N, 32]
    int*   __restrict__ outidx,         // [3, N] (int32 bit pattern in d_out tail)
    int nrows)
{
    extern __shared__ char smem[];
    float*     cwf  = (float*)(smem + OFF_CWF);
    _Float16*  cwh  = (_Float16*)(smem + OFF_CWH);
    float*     wsq  = (float*)(smem + OFF_WSQ);
    unsigned*  iscr = (unsigned*)(smem + OFF_IDX);

    const int tid = threadIdx.x;

    // ---- Stage f32 codebooks into LDS (24576 floats, 24 float4 per thread) ----
    {
        const float4* src = (const float4*)cb;
        float4*       dst = (float4*)cwf;
        #pragma unroll
        for (int i = 0; i < (LAYERS * KCW * DIM / 4) / 256; ++i)
            dst[tid + i * 256] = src[tid + i * 256];
    }
    __syncthreads();

    // ---- Build f16 copy + (||W||^2 + bias): 768 rows / 256 threads ----
    for (int row = tid; row < LAYERS * KCW; row += 256) {
        const float* p  = cwf + row * DIM;
        _Float16*    pb = cwh + row * DIM;
        float s = 0.0f;
        #pragma unroll
        for (int d = 0; d < DIM; ++d) {
            float v = p[d];
            s += v * v;
            pb[d] = (_Float16)v;
        }
        wsq[row] = s + DIST_BIAS;
    }
    __syncthreads();

    const int lane = tid & 31;
    const int wid  = tid >> 5;
    const int half = lane >> 4;   // 16-lane half
    const int lcol = lane & 15;   // A-row / B-col / C-col index
    const int row0 = blockIdx.x * ROWS_PER_WG + wid * ROWS_PER_WAVE;
    const size_t myrow = (size_t)(row0 + lcol);

    // ---- Load 16 rows of z into the WMMA A-fragment f32 layout ----
    // Lane (half,lcol): row=lcol, k in {8h..8h+7} (r[0..7]) and
    // {16+8h..23+8h} (r[8..15]); wave covers the full 16x128B tile.
    float r[16], q[16];
    {
        const float4* zp = (const float4*)(z + myrow * DIM);
        float4 a0 = zp[2 * half + 0];
        float4 a1 = zp[2 * half + 1];
        float4 a2 = zp[4 + 2 * half + 0];
        float4 a3 = zp[4 + 2 * half + 1];
        r[0]=a0.x;  r[1]=a0.y;  r[2]=a0.z;  r[3]=a0.w;
        r[4]=a1.x;  r[5]=a1.y;  r[6]=a1.z;  r[7]=a1.w;
        r[8]=a2.x;  r[9]=a2.y;  r[10]=a2.z; r[11]=a2.w;
        r[12]=a3.x; r[13]=a3.y; r[14]=a3.z; r[15]=a3.w;
        #pragma unroll
        for (int e = 0; e < 16; ++e) q[e] = 0.0f;
    }

    unsigned* widx = iscr + wid * ROWS_PER_WAVE;  // 16 slots, one per row

    #pragma unroll
    for (int layer = 0; layer < LAYERS; ++layer) {
        // A fragment: per-lane f32 -> f16 convert, no shuffles.
        v16h A;
        #pragma unroll
        for (int e = 0; e < 16; ++e) A[e] = (_Float16)r[e];

        const _Float16* cwhL = cwh + layer * KCW * DIM;
        const float*    wsqL = wsq + layer * KCW;

        // Per-row min slots (packed key = dist bits | codeword index).
        if (half == 0) widx[lcol] = 0xFFFFFFFFu;

        unsigned best[8];
        #pragma unroll
        for (int j = 0; j < 8; ++j) best[j] = 0xFFFFFFFFu;

        // Software-pipelined over 16 codeword tiles (256 codewords).
        // B fragment (32x16 f16): lane element e <-> K = 8*half + e (e<8),
        // 16 + 8*half + (e-8) (e>=8) of codeword c = 16t + lcol : two 16B LDS loads.
        v16h B0 = __builtin_shufflevector(*(const v8h*)(cwhL + (0 * 16 + lcol) * DIM + 8 * half), *(const v8h*)(cwhL + (0 * 16 + lcol) * DIM + 16 + 8 * half), 0, 1, 2, 3, 4, 5, 6, 7, 8, 9, 10, 11, 12, 13, 14, 15);
        v8f acc0 = {};
        acc0 = __builtin_amdgcn_wmma_f32_16x16x32_f16(
            false, A, false, B0, (short)0, acc0, false, false);

        #pragma unroll
        for (int t = 0; t < 16; ++t) {
            v8f accn = {};
            if (t + 1 < 16) {
                v16h Bn = __builtin_shufflevector(*(const v8h*)(cwhL + ((t + 1) * 16 + lcol) * DIM + 8 * half), *(const v8h*)(cwhL + ((t + 1) * 16 + lcol) * DIM + 16 + 8 * half), 0, 1, 2, 3, 4, 5, 6, 7, 8, 9, 10, 11, 12, 13, 14, 15);
                accn = __builtin_amdgcn_wmma_f32_16x16x32_f16(
                    false, A, false, Bn, (short)0, accn, false, false);
            }
            const float    wv = wsqL[t * 16 + lcol];       // ||W||^2 + bias
            const unsigned cu = (unsigned)(t * 16 + lcol); // codeword index (<256)
            #pragma unroll
            for (int j = 0; j < 8; ++j) {
                float d = __builtin_fmaf(-2.0f, acc0[j], wv);   // > 0 by bias
                unsigned key = (__float_as_uint(d) & 0xFFFFFF00u) | cu;
                best[j] = best[j] < key ? best[j] : key;        // v_min_u32
            }
            acc0 = accn;
        }

        // Per-row argmin tournament on the DS pipe: lane's acc rows are
        // m = j + 8*half; ds_min_u32 into the row's slot. min on packed keys
        // tie-breaks to the smallest codeword index (== jnp.argmin).
        #pragma unroll
        for (int j = 0; j < 8; ++j)
            atomicMin(&widx[8 * half + j], best[j]);
        __asm__ volatile("s_wait_dscnt 0" ::: "memory");  // same-wave DS RAW

        const unsigned keymin = widx[lcol];               // this lane's own row
        const int myIdx = (int)(keymin & 0xFFu);

        if (half == 0)
            outidx[(size_t)layer * nrows + myrow] = myIdx;

        // Subtract f32 codeword, accumulate quantized (same per-lane k layout).
        const float4* c4 = (const float4*)(cwf + (layer * KCW + myIdx) * DIM);
        float4 c0 = c4[2 * half + 0];
        float4 c1 = c4[2 * half + 1];
        float4 c2 = c4[4 + 2 * half + 0];
        float4 c3 = c4[4 + 2 * half + 1];
        r[0]-=c0.x;  q[0]+=c0.x;  r[1]-=c0.y;  q[1]+=c0.y;
        r[2]-=c0.z;  q[2]+=c0.z;  r[3]-=c0.w;  q[3]+=c0.w;
        r[4]-=c1.x;  q[4]+=c1.x;  r[5]-=c1.y;  q[5]+=c1.y;
        r[6]-=c1.z;  q[6]+=c1.z;  r[7]-=c1.w;  q[7]+=c1.w;
        r[8]-=c2.x;  q[8]+=c2.x;  r[9]-=c2.y;  q[9]+=c2.y;
        r[10]-=c2.z; q[10]+=c2.z; r[11]-=c2.w; q[11]+=c2.w;
        r[12]-=c3.x; q[12]+=c3.x; r[13]-=c3.y; q[13]+=c3.y;
        r[14]-=c3.z; q[14]+=c3.z; r[15]-=c3.w; q[15]+=c3.w;
    }

    // ---- Store quantized result (coalesced, same tile layout as the load) ----
    {
        float4* op = (float4*)(outq + myrow * DIM);
        op[2 * half + 0]     = make_float4(q[0],  q[1],  q[2],  q[3]);
        op[2 * half + 1]     = make_float4(q[4],  q[5],  q[6],  q[7]);
        op[4 + 2 * half + 0] = make_float4(q[8],  q[9],  q[10], q[11]);
        op[4 + 2 * half + 1] = make_float4(q[12], q[13], q[14], q[15]);
    }
}

extern "C" void kernel_launch(void* const* d_in, const int* in_sizes, int n_in,
                              void* d_out, int out_size, void* d_ws, size_t ws_size,
                              hipStream_t stream) {
    const float* z  = (const float*)d_in[0];   // [N, 32] f32
    const float* cb = (const float*)d_in[1];   // [3, 256, 32] f32
    const int nrows = in_sizes[0] / DIM;       // N = 524288

    float* outq   = (float*)d_out;                       // [N, 32] f32
    int*   outidx = (int*)(outq + (size_t)nrows * DIM);  // [3, N] i32 (bit-cast tail)

    const int blocks = nrows / ROWS_PER_WG;    // 4096 for N = 524288
    rq_vae_wmma_kernel<<<blocks, 256, SMEM_BYTES, stream>>>(z, cb, outq, outidx, nrows);
}
